// CustomContextEncoderForQG_5849745457187
// MI455X (gfx1250) — matrix operand order repaired and checked by reading, not compiled, not run on hardware
//
#include <hip/hip_runtime.h>
#include <hip/hip_bf16.h>

// ---------------------------------------------------------------------------
// Types / WMMA helpers (gfx1250, wave32)
// ---------------------------------------------------------------------------
typedef __attribute__((ext_vector_type(16))) _Float16 v16h;
typedef __attribute__((ext_vector_type(8)))  _Float16 v8h;
typedef __attribute__((ext_vector_type(8)))  float    v8f;

// Build a v16h from two 16-byte naturally-aligned chunks.
__device__ __forceinline__ v16h ld16(const _Float16* p0, const _Float16* p1) {
    v8h lo = *(const v8h*)p0;
    v8h hi = *(const v8h*)p1;
    return __builtin_shufflevector(lo, hi, 0,1,2,3,4,5,6,7,8,9,10,11,12,13,14,15);
}

__device__ __forceinline__ v8f wmma_f16(v16h a, v16h b, v8f c) {
    // v_wmma_f32_16x16x32_f16
    return __builtin_amdgcn_wmma_f32_16x16x32_f16(false, a, false, b,
                                                  (short)0, c, false, false);
}

// ---------------------------------------------------------------------------
// Elementwise converts / packing
// ---------------------------------------------------------------------------
__global__ void f32_to_f16_kernel(const float* __restrict__ in,
                                  _Float16* __restrict__ out, long n) {
    long i = (long)blockIdx.x * 256 + threadIdx.x;
    if (i < n) out[i] = (_Float16)in[i];
}

// v[m=b*512+s][n=h*128+d]  ->  vt[((b*10+h)*128+d)*512 + s]
__global__ void transpose_v_kernel(const _Float16* __restrict__ v,
                                   _Float16* __restrict__ vt) {
    long idx = (long)blockIdx.x * 256 + threadIdx.x;
    if (idx >= 8192L * 1280L) return;
    int m = (int)(idx / 1280), n = (int)(idx % 1280);
    int b = m >> 9, s = m & 511, h = n >> 7, d = n & 127;
    vt[(((long)((b * 10 + h) * 128 + d)) << 9) + s] = v[idx];
}

// ---------------------------------------------------------------------------
// GEMM: out[M][N](f16) = A[M][K](f16) x W[N][K]^T(f16) + bias[N](f32)
// Block: 128 threads = 4 waves. Wave computes a 32x64 strip (2 M-tiles x
// 4 N-tiles = 8 WMMA per K-step), software-pipelined: next K-step's 6
// fragments load while the current 8 WMMAs issue.  Grid: (N/256, M/32).
// ---------------------------------------------------------------------------
__global__ __launch_bounds__(128) void gemm_f16_kernel(
    const _Float16* __restrict__ A, const _Float16* __restrict__ W,
    const float* __restrict__ bias, _Float16* __restrict__ out,
    int M, int N, int K) {
    const int lane = threadIdx.x & 31, wave = threadIdx.x >> 5;
    const int lo = lane & 15, hi = lane >> 4;
    const int m0 = blockIdx.y * 32;
    const int n0 = (blockIdx.x * 16 + wave * 4) * 16;

    const _Float16* A0 = A + (long)(m0 + lo) * K + hi * 8;        // a[0..7], +16 for a[8..15]
    const _Float16* A1 = A + (long)(m0 + 16 + lo) * K + hi * 8;
    const _Float16* W0 = W + (long)(n0 + 0 * 16 + lo) * K + hi * 8;
    const _Float16* W1 = W + (long)(n0 + 1 * 16 + lo) * K + hi * 8;
    const _Float16* W2 = W + (long)(n0 + 2 * 16 + lo) * K + hi * 8;
    const _Float16* W3 = W + (long)(n0 + 3 * 16 + lo) * K + hi * 8;

    v8f acc[8] = {};
    v16h a0 = ld16(A0, A0 + 16);
    v16h a1 = ld16(A1, A1 + 16);
    v16h b0 = ld16(W0, W0 + 16);
    v16h b1 = ld16(W1, W1 + 16);
    v16h b2 = ld16(W2, W2 + 16);
    v16h b3 = ld16(W3, W3 + 16);

    for (int k = 32; k < K; k += 32) {
        v16h na0 = ld16(A0 + k, A0 + k + 16);
        v16h na1 = ld16(A1 + k, A1 + k + 16);
        v16h nb0 = ld16(W0 + k, W0 + k + 16);
        v16h nb1 = ld16(W1 + k, W1 + k + 16);
        v16h nb2 = ld16(W2 + k, W2 + k + 16);
        v16h nb3 = ld16(W3 + k, W3 + k + 16);
        acc[0] = wmma_f16(a0, b0, acc[0]);
        acc[1] = wmma_f16(a0, b1, acc[1]);
        acc[2] = wmma_f16(a0, b2, acc[2]);
        acc[3] = wmma_f16(a0, b3, acc[3]);
        acc[4] = wmma_f16(a1, b0, acc[4]);
        acc[5] = wmma_f16(a1, b1, acc[5]);
        acc[6] = wmma_f16(a1, b2, acc[6]);
        acc[7] = wmma_f16(a1, b3, acc[7]);
        a0 = na0; a1 = na1;
        b0 = nb0; b1 = nb1; b2 = nb2; b3 = nb3;
    }
    acc[0] = wmma_f16(a0, b0, acc[0]);
    acc[1] = wmma_f16(a0, b1, acc[1]);
    acc[2] = wmma_f16(a0, b2, acc[2]);
    acc[3] = wmma_f16(a0, b3, acc[3]);
    acc[4] = wmma_f16(a1, b0, acc[4]);
    acc[5] = wmma_f16(a1, b1, acc[5]);
    acc[6] = wmma_f16(a1, b2, acc[6]);
    acc[7] = wmma_f16(a1, b3, acc[7]);

#pragma unroll
    for (int mt = 0; mt < 2; ++mt)
#pragma unroll
        for (int t = 0; t < 4; ++t) {
            const int n = n0 + t * 16 + lo;
            const float bv = bias ? bias[n] : 0.f;
#pragma unroll
            for (int r = 0; r < 8; ++r) {
                long m = (long)m0 + mt * 16 + hi * 8 + r;  // C/D: VGPR r -> row r+8*hi
                out[m * N + n] = (_Float16)(acc[mt * 4 + t][r] + bv);
            }
        }
}

// ---------------------------------------------------------------------------
// BiLSTM scan.  One block per direction, 640 threads = 20 waves.
// Wave owns 2 h-column tiles; for each, all 4 gate tiles (i,f,g,o) so the
// cell update never crosses waves.  h carried in f32 regs + f16 LDS copy
// (double buffered, row-padded +8 f16 to spread LDS banks). c in regs only.
// All 8 B fragments are loaded into distinct registers before the 8-WMMA
// chain so later loads overlap earlier WMMAs (partial loadcnt waits).
// ---------------------------------------------------------------------------
#define HH 640
#define HP (HH + 8)

__global__ __launch_bounds__(640) void lstm_scan_kernel(
    const _Float16* __restrict__ xp,    // [8192][5120], dir at +d*2560
    const _Float16* __restrict__ whh,   // [2][2560][640]
    const int* __restrict__ lengths,    // [16]
    _Float16* __restrict__ out16,       // [8192][1280]
    float* __restrict__ out32) {        // optional (layer 1 -> d_out)
    const int d = blockIdx.x;
    const bool rev = (d == 1);
    const int lane = threadIdx.x & 31, wave = threadIdx.x >> 5;
    const int lo = lane & 15, hi = lane >> 4;

    __shared__ __align__(16) _Float16 hbuf[2][16][HP];
    __shared__ int len_s[16];

    if (threadIdx.x < 16) len_s[threadIdx.x] = lengths[threadIdx.x];
    for (int i = threadIdx.x; i < 2 * 16 * HP; i += 640)
        ((_Float16*)hbuf)[i] = (_Float16)0.f;
    __syncthreads();

    v8f c_reg[2] = {}, h_reg[2] = {};
    int lenr[8];
#pragma unroll
    for (int r = 0; r < 8; ++r) lenr[r] = len_s[hi * 8 + r];

    // Per-wave B pointers (8 gate tiles), K-contiguous rows of Whh.
    const _Float16* whd = whh + (long)d * 2560 * HH;
    const _Float16* Bp[8];
#pragma unroll
    for (int tt = 0; tt < 2; ++tt)
#pragma unroll
        for (int g = 0; g < 4; ++g) {
            const int ncol = (wave * 2 + tt) * 16 + lo;
            Bp[tt * 4 + g] = whd + (long)(g * HH + ncol) * HH + hi * 8;
        }

    for (int step = 0; step < 512; ++step) {
        const int t = rev ? 511 - step : step;
        const int buf = step & 1;

        v8f acc[8] = {};   // [tile(2)][gate(4)]
        for (int kk = 0; kk < HH; kk += 32) {
            const _Float16* ap = &hbuf[buf][lo][kk + hi * 8];
            v16h a = ld16(ap, ap + 16);
            v16h bfr[8];
#pragma unroll
            for (int i = 0; i < 8; ++i)
                bfr[i] = ld16(Bp[i] + kk, Bp[i] + kk + 16);
#pragma unroll
            for (int i = 0; i < 8; ++i)
                acc[i] = wmma_f16(a, bfr[i], acc[i]);
        }

#pragma unroll
        for (int tt = 0; tt < 2; ++tt) {
            const int ncol = (wave * 2 + tt) * 16 + lo;
#pragma unroll
            for (int r = 0; r < 8; ++r) {
                const int m = hi * 8 + r;            // batch row
                const long xb = ((long)m * 512 + t) * 5120 + (long)d * 2560;
                float gi = acc[tt * 4 + 0][r] + (float)xp[xb + 0 * HH + ncol];
                float gf = acc[tt * 4 + 1][r] + (float)xp[xb + 1 * HH + ncol];
                float gg = acc[tt * 4 + 2][r] + (float)xp[xb + 2 * HH + ncol];
                float go = acc[tt * 4 + 3][r] + (float)xp[xb + 3 * HH + ncol];
                float iv = 1.f / (1.f + __expf(-gi));
                float fv = 1.f / (1.f + __expf(-gf));
                float gv = tanhf(gg);
                float ov = 1.f / (1.f + __expf(-go));
                float c2 = fv * c_reg[tt][r] + iv * gv;
                float h2 = ov * tanhf(c2);
                bool valid = t < lenr[r];
                float cn = valid ? c2 : c_reg[tt][r];
                float hn = valid ? h2 : h_reg[tt][r];
                c_reg[tt][r] = cn;
                h_reg[tt][r] = hn;
                float yv = valid ? h2 : 0.f;
                const long ob = ((long)m * 512 + t) * 1280 + (long)d * HH + ncol;
                out16[ob] = (_Float16)yv;
                if (out32) out32[ob] = yv;
                hbuf[buf ^ 1][m][ncol] = (_Float16)hn;
            }
        }
        __syncthreads();   // publish h for next step
    }
}

// ---------------------------------------------------------------------------
// Attention.  Grid: (32 q-tiles, 160 (b,head)).  256 threads = 8 waves.
//   scores(16x512) = Q(16x128) K^T / sqrt(128) + mask   (WMMA, 4 tiles/wave,
//       pipelined across the 4 K-steps)
//   softmax rows in LDS (wave shuffle reductions)
//   ctx(16x128) = P(16x512, f16 LDS) x V^T-layout vt    (WMMA, pipelined)
//   d_out += ctx  (d_out already holds masked h)
// ---------------------------------------------------------------------------
__global__ __launch_bounds__(256) void attention_kernel(
    const _Float16* __restrict__ q, const _Float16* __restrict__ k,
    const _Float16* __restrict__ vt, const float* __restrict__ mask,
    float* __restrict__ out) {
    const int qt = blockIdx.x;               // query tile 0..31
    const int bh = blockIdx.y;               // 0..159
    const int b = bh / 10, h = bh % 10;
    const int lane = threadIdx.x & 31, wave = threadIdx.x >> 5;
    const int lo = lane & 15, hi = lane >> 4;

    __shared__ float s_lds[16][513];
    __shared__ __align__(16) _Float16 p_lds[16][520];
    const float scale = 0.08838834764831845f;   // 1/sqrt(128)

    const _Float16* Qp =
        q + ((long)(b * 512 + qt * 16 + lo)) * 1280 + h * 128 + hi * 8;
    const _Float16* Kp[4];
#pragma unroll
    for (int j = 0; j < 4; ++j) {
        const int s = (wave + 8 * j) * 16 + lo;
        Kp[j] = k + ((long)(b * 512 + s)) * 1280 + h * 128 + hi * 8;
    }

    v8f acc[4] = {};
    v16h a = ld16(Qp, Qp + 16);
    v16h b0 = ld16(Kp[0], Kp[0] + 16);
    v16h b1 = ld16(Kp[1], Kp[1] + 16);
    v16h b2 = ld16(Kp[2], Kp[2] + 16);
    v16h b3 = ld16(Kp[3], Kp[3] + 16);
    for (int kk = 32; kk < 128; kk += 32) {
        v16h na = ld16(Qp + kk, Qp + kk + 16);
        v16h nb0 = ld16(Kp[0] + kk, Kp[0] + kk + 16);
        v16h nb1 = ld16(Kp[1] + kk, Kp[1] + kk + 16);
        v16h nb2 = ld16(Kp[2] + kk, Kp[2] + kk + 16);
        v16h nb3 = ld16(Kp[3] + kk, Kp[3] + kk + 16);
        acc[0] = wmma_f16(a, b0, acc[0]);
        acc[1] = wmma_f16(a, b1, acc[1]);
        acc[2] = wmma_f16(a, b2, acc[2]);
        acc[3] = wmma_f16(a, b3, acc[3]);
        a = na; b0 = nb0; b1 = nb1; b2 = nb2; b3 = nb3;
    }
    acc[0] = wmma_f16(a, b0, acc[0]);
    acc[1] = wmma_f16(a, b1, acc[1]);
    acc[2] = wmma_f16(a, b2, acc[2]);
    acc[3] = wmma_f16(a, b3, acc[3]);

#pragma unroll
    for (int j = 0; j < 4; ++j) {
        const int s = (wave + 8 * j) * 16 + lo;
        const float mk = mask[b * 512 + s];
#pragma unroll
        for (int r = 0; r < 8; ++r)
            s_lds[hi * 8 + r][s] = acc[j][r] * scale + mk;
    }
    __syncthreads();

    // softmax: wave handles rows 2*wave, 2*wave+1
    for (int rr = 0; rr < 2; ++rr) {
        const int row = wave * 2 + rr;
        float mx = -3.0e38f;
        for (int i = lane; i < 512; i += 32) mx = fmaxf(mx, s_lds[row][i]);
#pragma unroll
        for (int off = 16; off > 0; off >>= 1)
            mx = fmaxf(mx, __shfl_xor(mx, off, 32));
        float sum = 0.f;
        for (int i = lane; i < 512; i += 32) {
            float e = __expf(s_lds[row][i] - mx);
            s_lds[row][i] = e;
            sum += e;
        }
#pragma unroll
        for (int off = 16; off > 0; off >>= 1) sum += __shfl_xor(sum, off, 32);
        const float inv = 1.f / sum;
        for (int i = lane; i < 512; i += 32)
            p_lds[row][i] = (_Float16)(s_lds[row][i] * inv);
    }
    __syncthreads();

    // ctx: wave owns one 16-wide d-tile (8 waves x 16 = 128), pipelined
    const _Float16* Vp = vt + ((long)(bh * 128 + wave * 16 + lo)) * 512 + hi * 8;
    v8f cacc = {};
    v16h pa = ld16(&p_lds[lo][hi * 8], &p_lds[lo][hi * 8 + 16]);
    v16h vb = ld16(Vp, Vp + 16);
    for (int kk = 32; kk < 512; kk += 32) {
        v16h npa = ld16(&p_lds[lo][kk + hi * 8], &p_lds[lo][kk + hi * 8 + 16]);
        v16h nvb = ld16(Vp + kk, Vp + kk + 16);
        cacc = wmma_f16(pa, vb, cacc);
        pa = npa; vb = nvb;
    }
    cacc = wmma_f16(pa, vb, cacc);

#pragma unroll
    for (int r = 0; r < 8; ++r) {
        const long o = ((long)(b * 512 + qt * 16 + hi * 8 + r)) * 1280 +
                       h * 128 + wave * 16 + lo;
        out[o] += cacc[r];
    }
}

// ---------------------------------------------------------------------------
// Host side
// ---------------------------------------------------------------------------
extern "C" void kernel_launch(void* const* d_in, const int* in_sizes, int n_in,
                              void* d_out, int out_size, void* d_ws, size_t ws_size,
                              hipStream_t stream) {
    (void)in_sizes; (void)n_in; (void)out_size; (void)ws_size;
    const float* x    = (const float*)d_in[0];
    const float* mask = (const float*)d_in[1];
    const int*   lens = (const int*)d_in[2];
    const float* Wih0 = (const float*)d_in[3];
    const float* Whh0 = (const float*)d_in[4];
    const float* b0   = (const float*)d_in[5];
    const float* Wih1 = (const float*)d_in[6];
    const float* Whh1 = (const float*)d_in[7];
    const float* b1   = (const float*)d_in[8];
    const float* Wq   = (const float*)d_in[9];
    const float* bq   = (const float*)d_in[10];
    const float* Wk   = (const float*)d_in[11];
    const float* bk   = (const float*)d_in[12];
    const float* Wv   = (const float*)d_in[13];
    const float* bv   = (const float*)d_in[14];

    char* ws = (char*)d_ws;
    size_t off = 0;
    auto alloc = [&](long elems) {
        _Float16* p = (_Float16*)(ws + off);
        off = (off + (size_t)elems * sizeof(_Float16) + 255) & ~(size_t)255;
        return p;
    };
    const int M = 8192;
    _Float16* x16    = alloc((long)M * 768);
    _Float16* wih0h  = alloc(5120L * 768);
    _Float16* whh0h  = alloc(5120L * 640);
    _Float16* wih1h  = alloc(5120L * 1280);
    _Float16* whh1h  = alloc(5120L * 640);
    _Float16* wqh    = alloc(1280L * 1280);
    _Float16* wkh    = alloc(1280L * 1280);
    _Float16* wvh    = alloc(1280L * 1280);
    _Float16* xp16   = alloc((long)M * 5120);   // reused as vt after scans
    _Float16* y016   = alloc((long)M * 1280);   // reused as q after layer-1 gemm
    _Float16* h116   = alloc((long)M * 1280);
    _Float16* k16    = alloc((long)M * 1280);
    _Float16* v16p   = alloc((long)M * 1280);
    _Float16* q16 = y016;
    _Float16* vt  = xp16;

    auto conv = [&](const float* in, _Float16* outp, long n) {
        f32_to_f16_kernel<<<dim3((unsigned)((n + 255) / 256)), 256, 0, stream>>>(in, outp, n);
    };
    conv(x,    x16,   (long)M * 768);
    conv(Wih0, wih0h, 5120L * 768);
    conv(Whh0, whh0h, 5120L * 640);
    conv(Wih1, wih1h, 5120L * 1280);
    conv(Whh1, whh1h, 5120L * 640);
    conv(Wq,   wqh,   1280L * 1280);
    conv(Wk,   wkh,   1280L * 1280);
    conv(Wv,   wvh,   1280L * 1280);

    // Layer 0: xp = x @ Wih0^T + b0  (both directions, N=5120) ; then scan
    gemm_f16_kernel<<<dim3(5120 / 256, M / 32), 128, 0, stream>>>(
        x16, wih0h, b0, xp16, M, 5120, 768);
    lstm_scan_kernel<<<2, 640, 0, stream>>>(xp16, whh0h, lens, y016, nullptr);

    // Layer 1
    gemm_f16_kernel<<<dim3(5120 / 256, M / 32), 128, 0, stream>>>(
        y016, wih1h, b1, xp16, M, 5120, 1280);
    lstm_scan_kernel<<<2, 640, 0, stream>>>(xp16, whh1h, lens, h116,
                                            (float*)d_out);

    // q, k, v projections
    gemm_f16_kernel<<<dim3(1280 / 256, M / 32), 128, 0, stream>>>(
        h116, wqh, bq, q16, M, 1280, 1280);
    gemm_f16_kernel<<<dim3(1280 / 256, M / 32), 128, 0, stream>>>(
        h116, wkh, bk, k16, M, 1280, 1280);
    gemm_f16_kernel<<<dim3(1280 / 256, M / 32), 128, 0, stream>>>(
        h116, wvh, bv, v16p, M, 1280, 1280);
    transpose_v_kernel<<<dim3((unsigned)(((long)M * 1280 + 255) / 256)), 256, 0,
                         stream>>>(v16p, vt);

    // attention + residual accumulate into d_out
    attention_kernel<<<dim3(32, 160), 256, 0, stream>>>(q16, k16, vt, mask,
                                                        (float*)d_out);
}
